// SARANDeep_80118319940147
// MI455X (gfx1250) — compile-verified, matrix operand order checked by reading, not run on hardware
//
#include <hip/hip_runtime.h>
#include <hip/hip_bf16.h>

typedef __attribute__((ext_vector_type(16))) _Float16 v16h;
typedef __attribute__((ext_vector_type(8)))  _Float16 v8h;
typedef __attribute__((ext_vector_type(8)))  float    v8f;

#define VOCAB_SZ 50257
#define C_EMBD   1024
#define N_LAYER  3
#define B_SZ     2
#define T_SZ     2048
#define BT_SZ    (B_SZ * T_SZ)

// ---------------------------------------------------------------------------
// WMMA fragment helpers.  16-bit A/B 16x32 layout (ISA 7.12.2): lanes 0-15
// hold row M=lane with K={0..7} in v0..v3 and K={16..23} in v4..v7; lanes
// 16-31 hold the same rows with K={8..15} and {24..31}.  Each lane therefore
// loads two contiguous 16-byte chunks at k0 + half*8 and k0 + 16 + half*8.
// ---------------------------------------------------------------------------
__device__ __forceinline__ v16h pack16(v8h lo, v8h hi) {
  v16h f;
#pragma unroll
  for (int i = 0; i < 8; ++i) { f[i] = lo[i]; f[i + 8] = hi[i]; }
  return f;
}

// 64x64 macro-tile (4x4 WMMA tiles) of A(MxK) * B(NxK)^T.
// 8 fragment loads feed 16 WMMAs per K-step of 32 (0.5 b128-pair per WMMA).
__device__ __forceinline__ void gemm_block44(const _Float16* __restrict__ A,
                                             const _Float16* __restrict__ Bm,
                                             int lda, int ldb, int K,
                                             int row0, int col0, int lane,
                                             int rmaxB, v8f (&acc)[4][4]) {
  const int rA = lane & 15;
  const int half8 = (lane >> 4) * 8;
  const _Float16* pa[4];
  const _Float16* pb[4];
#pragma unroll
  for (int i = 0; i < 4; ++i)
    pa[i] = A + (size_t)(row0 + 16 * i + rA) * lda + half8;
#pragma unroll
  for (int j = 0; j < 4; ++j) {
    int r = col0 + 16 * j + rA;
    if (r > rmaxB) r = rmaxB;                  // ragged-N clamp (logits)
    pb[j] = Bm + (size_t)r * ldb + half8;
  }
  for (int k0 = 0; k0 < K; k0 += 32) {
    v16h a[4], b[4];
#pragma unroll
    for (int i = 0; i < 4; ++i) {
      a[i] = pack16(*(const v8h*)(pa[i] + k0), *(const v8h*)(pa[i] + k0 + 16));
      __builtin_prefetch((const void*)(pa[i] + k0 + 32), 0, 3);  // speculative
    }
#pragma unroll
    for (int j = 0; j < 4; ++j) {
      b[j] = pack16(*(const v8h*)(pb[j] + k0), *(const v8h*)(pb[j] + k0 + 16));
      __builtin_prefetch((const void*)(pb[j] + k0 + 32), 0, 3);
    }
#pragma unroll
    for (int i = 0; i < 4; ++i)
#pragma unroll
      for (int j = 0; j < 4; ++j)
        acc[i][j] = __builtin_amdgcn_wmma_f32_16x16x32_f16(
            false, a[i], false, b[j], (short)0, acc[i][j], false, false);
  }
}

// ---------------------------------------------------------------------------
// Elementwise / prep kernels
// ---------------------------------------------------------------------------
__global__ void k_cvt_f16(const float* __restrict__ s, _Float16* __restrict__ d, int n) {
  int i = (blockIdx.x * blockDim.x + threadIdx.x) * 4;
  if (i >= n) return;
  float4 v = *(const float4*)(s + i);
  d[i + 0] = (_Float16)v.x;
  d[i + 1] = (_Float16)v.y;
  d[i + 2] = (_Float16)v.z;
  d[i + 3] = (_Float16)v.w;
}

__global__ void k_embed(const int* __restrict__ idx,
                        const float* __restrict__ w_embed,
                        const float* __restrict__ w_pos,
                        float* __restrict__ x32, _Float16* __restrict__ x16) {
  int row = blockIdx.x;                 // 0 .. B*T-1
  int t = row & (T_SZ - 1);
  int tok = idx[row];
  const float* e = w_embed + (size_t)tok * C_EMBD;
  const float* p = w_pos + (size_t)t * C_EMBD;
  for (int c = threadIdx.x; c < C_EMBD; c += blockDim.x) {
    float v = e[c] + p[c];
    x32[(size_t)row * C_EMBD + c] = v;
    x16[(size_t)row * C_EMBD + c] = (_Float16)v;
  }
}

// ---------------------------------------------------------------------------
// GEMM kernels (64x64 macro-tile per wave, 8 waves / 256-thread block)
// ---------------------------------------------------------------------------

// Y(MxN) f16 = X(MxK) @ W(NxK)^T   (q/k/v projections)
__global__ __launch_bounds__(256) void k_gemm_nt_f16out(
    const _Float16* __restrict__ A, const _Float16* __restrict__ Bm,
    _Float16* __restrict__ Y, int M, int N, int K) {
  int wave = threadIdx.x >> 5, lane = threadIdx.x & 31;
  int tile = blockIdx.x * 8 + wave;
  int mtM = M >> 6, mtN = N >> 6;
  if (tile >= mtM * mtN) return;
  int row0 = (tile % mtM) * 64;
  int col0 = (tile / mtM) * 64;
  v8f acc[4][4] = {};
  gemm_block44(A, Bm, K, K, K, row0, col0, lane, N - 1, acc);
  int m0 = row0 + (lane >> 4) * 8;
  int n0 = col0 + (lane & 15);
#pragma unroll
  for (int i = 0; i < 4; ++i)
#pragma unroll
    for (int j = 0; j < 4; ++j)
#pragma unroll
      for (int e = 0; e < 8; ++e)
        Y[(size_t)(m0 + 16 * i + e) * N + n0 + 16 * j] = (_Float16)acc[i][j][e];
}

// S[b,q,k] = (Q[b,q,:]·K[b,k,:]) * inv_scale, causal-masked with -inf.
__global__ __launch_bounds__(256) void k_scores(
    const _Float16* __restrict__ Q, const _Float16* __restrict__ Km,
    float* __restrict__ S, float inv_scale) {
  int b = blockIdx.y;
  int wave = threadIdx.x >> 5, lane = threadIdx.x & 31;
  int tile = blockIdx.x * 8 + wave;
  const int mtM = T_SZ >> 6;
  int row0 = (tile % mtM) * 64;
  int col0 = (tile / mtM) * 64;
  float* Sb = S + (size_t)b * T_SZ * T_SZ;
  int m0 = row0 + (lane >> 4) * 8;
  int n0 = col0 + (lane & 15);
  if (col0 > row0 + 63) {               // macro-tile fully above diagonal
#pragma unroll
    for (int i = 0; i < 4; ++i)
#pragma unroll
      for (int j = 0; j < 4; ++j)
#pragma unroll
        for (int e = 0; e < 8; ++e)
          Sb[(size_t)(m0 + 16 * i + e) * T_SZ + n0 + 16 * j] = -__builtin_inff();
    return;
  }
  v8f acc[4][4] = {};
  gemm_block44(Q + (size_t)b * T_SZ * C_EMBD, Km + (size_t)b * T_SZ * C_EMBD,
               C_EMBD, C_EMBD, C_EMBD, row0, col0, lane, T_SZ - 1, acc);
#pragma unroll
  for (int i = 0; i < 4; ++i)
#pragma unroll
    for (int j = 0; j < 4; ++j)
#pragma unroll
      for (int e = 0; e < 8; ++e) {
        int m = m0 + 16 * i + e, n = n0 + 16 * j;
        Sb[(size_t)m * T_SZ + n] =
            (n <= m) ? acc[i][j][e] * inv_scale : -__builtin_inff();
      }
}

// Row softmax (length T) -> f16 probabilities. One 256-thread block per row.
__global__ void k_softmax(const float* __restrict__ S, _Float16* __restrict__ P) {
  __shared__ float red[256];
  size_t row = blockIdx.x;
  const float* s = S + row * T_SZ;
  _Float16* p = P + row * T_SZ;
  int tid = threadIdx.x;
  float lmax = -__builtin_inff();
  for (int j = tid; j < T_SZ; j += 256) lmax = fmaxf(lmax, s[j]);
  red[tid] = lmax;
  __syncthreads();
  for (int st = 128; st > 0; st >>= 1) {
    if (tid < st) red[tid] = fmaxf(red[tid], red[tid + st]);
    __syncthreads();
  }
  float m = red[0];
  __syncthreads();
  float lsum = 0.f;
  for (int j = tid; j < T_SZ; j += 256) lsum += __expf(s[j] - m);
  red[tid] = lsum;
  __syncthreads();
  for (int st = 128; st > 0; st >>= 1) {
    if (tid < st) red[tid] += red[tid + st];
    __syncthreads();
  }
  float inv = 1.0f / red[0];
  for (int j = tid; j < T_SZ; j += 256) p[j] = (_Float16)(__expf(s[j] - m) * inv);
}

// Per-batch f16 transpose: VT[b][d*T + k] = V[b][k*C + d]  (LDS 32x33 tile)
__global__ void k_transpose_f16(const _Float16* __restrict__ V,
                                _Float16* __restrict__ VT) {
  __shared__ _Float16 tile[32][33];
  int b = blockIdx.z;
  int k0 = blockIdx.x * 32, d0 = blockIdx.y * 32;
  int tx = threadIdx.x & 31, ty = threadIdx.x >> 5;   // 256 thr: ty 0..7
  const _Float16* Vb = V + (size_t)b * T_SZ * C_EMBD;
  _Float16* VTb = VT + (size_t)b * C_EMBD * T_SZ;
#pragma unroll
  for (int i = 0; i < 32; i += 8)
    tile[ty + i][tx] = Vb[(size_t)(k0 + ty + i) * C_EMBD + d0 + tx];
  __syncthreads();
#pragma unroll
  for (int i = 0; i < 32; i += 8)
    VTb[(size_t)(d0 + ty + i) * T_SZ + k0 + tx] = tile[tx][ty + i];
}

// x += P @ V  using VT (per batch):  out[q,d] = sum_k P[q,k] * VT[d,k]
// Fused residual epilogue: update x_f32 and its f16 mirror in place.
__global__ __launch_bounds__(256) void k_attnv(
    const _Float16* __restrict__ P, const _Float16* __restrict__ VT,
    float* __restrict__ x32, _Float16* __restrict__ x16) {
  int b = blockIdx.y;
  int wave = threadIdx.x >> 5, lane = threadIdx.x & 31;
  int tile = blockIdx.x * 8 + wave;
  const int mtM = T_SZ >> 6;
  if (tile >= mtM * (C_EMBD >> 6)) return;
  int row0 = (tile % mtM) * 64;
  int col0 = (tile / mtM) * 64;
  v8f acc[4][4] = {};
  gemm_block44(P + (size_t)b * T_SZ * T_SZ, VT + (size_t)b * C_EMBD * T_SZ,
               T_SZ, T_SZ, T_SZ, row0, col0, lane, C_EMBD - 1, acc);
  int m0 = row0 + (lane >> 4) * 8;
  int n0 = col0 + (lane & 15);
#pragma unroll
  for (int i = 0; i < 4; ++i)
#pragma unroll
    for (int j = 0; j < 4; ++j)
#pragma unroll
      for (int e = 0; e < 8; ++e) {
        size_t idx = (size_t)(b * T_SZ + m0 + 16 * i + e) * C_EMBD + n0 + 16 * j;
        float v = x32[idx] + acc[i][j][e];
        x32[idx] = v;
        x16[idx] = (_Float16)v;
      }
}

// logits[m, v] = X[m,:] · Wo[v,:] + b_out[v]   (N ragged: 50257)
__global__ __launch_bounds__(256) void k_logits(
    const _Float16* __restrict__ X, const _Float16* __restrict__ Wo,
    const float* __restrict__ bias, float* __restrict__ out) {
  int wave = threadIdx.x >> 5, lane = threadIdx.x & 31;
  int tile = blockIdx.x * 8 + wave;
  const int mtM = BT_SZ >> 6;
  const int mtN = (VOCAB_SZ + 63) >> 6;              // 786
  if (tile >= mtM * mtN) return;
  int row0 = (tile % mtM) * 64;
  int col0 = (tile / mtM) * 64;
  v8f acc[4][4] = {};
  gemm_block44(X, Wo, C_EMBD, C_EMBD, C_EMBD, row0, col0, lane, VOCAB_SZ - 1, acc);
  int m0 = row0 + (lane >> 4) * 8;
#pragma unroll
  for (int j = 0; j < 4; ++j) {
    int n = col0 + 16 * j + (lane & 15);
    if (n < VOCAB_SZ) {
      float bv = bias[n];
#pragma unroll
      for (int i = 0; i < 4; ++i)
#pragma unroll
        for (int e = 0; e < 8; ++e)
          out[(size_t)(m0 + 16 * i + e) * VOCAB_SZ + n] = acc[i][j][e] + bv;
    }
  }
}

// ---------------------------------------------------------------------------
// Host launcher
// ---------------------------------------------------------------------------
extern "C" void kernel_launch(void* const* d_in, const int* in_sizes, int n_in,
                              void* d_out, int out_size, void* d_ws, size_t ws_size,
                              hipStream_t stream) {
  const int*   idx     = (const int*)d_in[0];
  const float* w_embed = (const float*)d_in[1];
  const float* w_pos   = (const float*)d_in[2];
  const float* wq      = (const float*)d_in[3];
  const float* wk      = (const float*)d_in[4];
  const float* wv      = (const float*)d_in[5];
  const float* w_out   = (const float*)d_in[6];
  const float* b_out   = (const float*)d_in[7];
  float* out = (float*)d_out;

  // Workspace layout (all offsets 256B-aligned)
  char* ws = (char*)d_ws;
  const size_t SZ_X32 = (size_t)BT_SZ * C_EMBD * 4;        // 16.8 MB
  const size_t SZ_X16 = (size_t)BT_SZ * C_EMBD * 2;        //  8.4 MB
  const size_t SZ_S   = (size_t)B_SZ * T_SZ * T_SZ * 4;    // 33.6 MB
  const size_t SZ_P   = (size_t)B_SZ * T_SZ * T_SZ * 2;    // 16.8 MB
  const size_t SZ_WL  = (size_t)N_LAYER * C_EMBD * C_EMBD * 2;
  const size_t SZ_WO  = (size_t)VOCAB_SZ * C_EMBD * 2;

  size_t off = 0;
  float*    x32  = (float*)(ws + off);    off += SZ_X32;
  _Float16* x16  = (_Float16*)(ws + off); off += SZ_X16;
  _Float16* q16  = (_Float16*)(ws + off); off += SZ_X16;
  _Float16* k16  = (_Float16*)(ws + off); off += SZ_X16;
  _Float16* v16_ = (_Float16*)(ws + off); off += SZ_X16;
  _Float16* vt16 = (_Float16*)(ws + off); off += SZ_X16;
  float*    S    = (float*)(ws + off);    off += SZ_S;
  _Float16* P16  = (_Float16*)(ws + off); off += SZ_P;
  _Float16* wq16 = (_Float16*)(ws + off); off += SZ_WL;
  _Float16* wk16 = (_Float16*)(ws + off); off += SZ_WL;
  _Float16* wv16 = (_Float16*)(ws + off); off += SZ_WL;
  _Float16* wo16 = (_Float16*)(ws + off); off += SZ_WO;

  // 1) one-time weight conversion to f16
  {
    int nw = N_LAYER * C_EMBD * C_EMBD;
    int blocks = (nw / 4 + 255) / 256;
    k_cvt_f16<<<blocks, 256, 0, stream>>>(wq, wq16, nw);
    k_cvt_f16<<<blocks, 256, 0, stream>>>(wk, wk16, nw);
    k_cvt_f16<<<blocks, 256, 0, stream>>>(wv, wv16, nw);
    int no = VOCAB_SZ * C_EMBD;
    k_cvt_f16<<<(no / 4 + 255) / 256, 256, 0, stream>>>(w_out, wo16, no);
  }

  // 2) embeddings
  k_embed<<<BT_SZ, 256, 0, stream>>>(idx, w_embed, w_pos, x32, x16);

  // 3) attention layers (macro-tiles are 64x64; 8 waves per block)
  const int mt_qkv = (BT_SZ / 64) * (C_EMBD / 64);             // 1024
  const int mt_sc  = (T_SZ / 64) * (T_SZ / 64);                // 1024 / batch
  const int mt_av  = (T_SZ / 64) * (C_EMBD / 64);              //  512 / batch
  for (int L = 0; L < N_LAYER; ++L) {
    const _Float16* wql = wq16 + (size_t)L * C_EMBD * C_EMBD;
    const _Float16* wkl = wk16 + (size_t)L * C_EMBD * C_EMBD;
    const _Float16* wvl = wv16 + (size_t)L * C_EMBD * C_EMBD;
    k_gemm_nt_f16out<<<mt_qkv / 8, 256, 0, stream>>>(x16, wql, q16, BT_SZ, C_EMBD, C_EMBD);
    k_gemm_nt_f16out<<<mt_qkv / 8, 256, 0, stream>>>(x16, wkl, k16, BT_SZ, C_EMBD, C_EMBD);
    k_gemm_nt_f16out<<<mt_qkv / 8, 256, 0, stream>>>(x16, wvl, v16_, BT_SZ, C_EMBD, C_EMBD);

    dim3 gs(mt_sc / 8, B_SZ);
    k_scores<<<gs, 256, 0, stream>>>(q16, k16, S, 1.0f / 32.0f);  // 1/sqrt(1024)

    k_softmax<<<BT_SZ, 256, 0, stream>>>(S, P16);

    dim3 gt(T_SZ / 32, C_EMBD / 32, B_SZ);
    k_transpose_f16<<<gt, 256, 0, stream>>>(v16_, vt16);

    dim3 ga(mt_av / 8, B_SZ);
    k_attnv<<<ga, 256, 0, stream>>>(P16, vt16, x32, x16);
  }

  // 4) logits
  const int mtN = (VOCAB_SZ + 63) / 64;                        // 786
  const int total = (BT_SZ / 64) * mtN;                        // 50304
  k_logits<<<(total + 7) / 8, 256, 0, stream>>>(x16, wo16, b_out, out);
}
